// Policy_Module2_86053964742746
// MI455X (gfx1250) — compile-verified, matrix operand order checked by reading, not run on hardware
//
#include <hip/hip_runtime.h>
#include <hip/hip_bf16.h>
#include <stdint.h>

typedef __attribute__((ext_vector_type(16))) _Float16 v16h;
typedef __attribute__((ext_vector_type(8)))  _Float16 h8;
typedef __attribute__((ext_vector_type(8)))  float    v8f;

#define EPS_GEN 1e-7f
#define LN_EPS  1e-5f

// ---------------------------------------------------------------------------
// WMMA helpers
// ---------------------------------------------------------------------------
__device__ __forceinline__ v8f wmma16(v16h a, v16h b, v8f c) {
    // emits v_wmma_f32_16x16x32_f16
    return __builtin_amdgcn_wmma_f32_16x16x32_f16(
        /*neg_a=*/false, a, /*neg_b=*/false, b,
        /*c_mod=*/(short)0, c, /*reuse_a=*/false, /*reuse_b=*/false);
}

// 16x32 f16 fragment load (A layout; B uses same pattern over columns of W^T).
// Row-major source, ld in halves. lanes 0-15: K [kb..kb+8)+[kb+16..kb+24),
// lanes 16-31: +8 halves shift, rows repeat.
__device__ __forceinline__ v16h load_frag(const _Float16* base, int ld,
                                          int rowbase, int kbase, int lane) {
    int row = rowbase + (lane & 15);
    int hi  = (lane >> 4) * 8;
    const h8* p0 = (const h8*)(base + (size_t)row * ld + kbase + hi);
    const h8* p1 = (const h8*)(base + (size_t)row * ld + kbase + 16 + hi);
    h8 x0 = *p0, x1 = *p1;
    v16h r;
#pragma unroll
    for (int i = 0; i < 8; i++) { r[i] = x0[i]; r[8 + i] = x1[i]; }
    return r;
}

// order-preserving float <-> uint map for atomic max on f32
__device__ __forceinline__ unsigned fenc(float f) {
    unsigned u = __float_as_uint(f);
    return (u & 0x80000000u) ? ~u : (u | 0x80000000u);
}
__device__ __forceinline__ float fdec(unsigned u) {
    return (u & 0x80000000u) ? __uint_as_float(u & 0x7FFFFFFFu)
                             : __uint_as_float(~u);
}

__device__ __forceinline__ float wave_sum(float v) {
#pragma unroll
    for (int off = 16; off > 0; off >>= 1) v += __shfl_xor(v, off, 32);
    return v;
}

// ---------------------------------------------------------------------------
// Weight prep: dst[j][k] = (f16) src[k*Nout + j]  (transpose + pad K)
// ---------------------------------------------------------------------------
__global__ void wconv_kernel(const float* __restrict__ src, _Float16* __restrict__ dst,
                             int K, int Nout, int padK) {
    int i = blockIdx.x * 256 + threadIdx.x;
    if (i >= Nout * padK) return;
    int j = i / padK, k = i - j * padK;
    float v = (k < K) ? src[k * Nout + j] : 0.f;
    dst[i] = (_Float16)v;
}

// ---------------------------------------------------------------------------
// Encoder: h[N,64] = x[N,16] @ W_enc + b_enc   (K padded 16->32)
// block = 128 threads (4 waves), 16 nodes / block, wave w -> 16-col tile w
// ---------------------------------------------------------------------------
__global__ void encoder_kernel(const float* __restrict__ x,
                               const _Float16* __restrict__ WencT, // [64][32]
                               const float* __restrict__ benc,
                               float* __restrict__ h, int N) {
    __shared__ __align__(16) _Float16 aLds[16][32];
    int tid = threadIdx.x, wave = tid >> 5, lane = tid & 31;
    int node0 = blockIdx.x * 16;
    for (int i = tid; i < 16 * 32; i += 128) {
        int r = i >> 5, k = i & 31;
        int node = node0 + r;
        float v = (k < 16 && node < N) ? x[(size_t)node * 16 + k] : 0.f;
        aLds[r][k] = (_Float16)v;
    }
    __syncthreads();
    v16h a = load_frag(&aLds[0][0], 32, 0, 0, lane);
    v16h b = load_frag(WencT, 32, wave * 16, 0, lane);
    v8f  c = {};
    c = wmma16(a, b, c);
    int col   = wave * 16 + (lane & 15);
    int rbase = node0 + (lane >> 4) * 8;
    float bias = benc[col];
    if (node0 + 16 <= N) {
#pragma unroll
        for (int r = 0; r < 8; r++)
            h[(size_t)(rbase + r) * 64 + col] = c[r] + bias;
    } else {
#pragma unroll
        for (int r = 0; r < 8; r++)
            if (rbase + r < N) h[(size_t)(rbase + r) * 64 + col] = c[r] + bias;
    }
}

// ---------------------------------------------------------------------------
// conv scratch init (vectorized; NH multiple of 4)
// ---------------------------------------------------------------------------
__global__ void init_kernel(uint4* __restrict__ m_u, float4* __restrict__ denom,
                            float4* __restrict__ agg, long long NH4) {
    long long g = (long long)blockIdx.x * 256 + threadIdx.x;
    if (g >= NH4) return;
    m_u[g]   = make_uint4(0u, 0u, 0u, 0u);
    denom[g] = make_float4(0.f, 0.f, 0.f, 0.f);
    agg[g]   = make_float4(0.f, 0.f, 0.f, 0.f);
}

// ---------------------------------------------------------------------------
// Edge pass 1: per-(dst,feature) running max of z = (relu(x[src])+eps)*t
// 4 features / thread: one float4 gather + 4 u32 atomic max
// ---------------------------------------------------------------------------
__global__ void edge_max_kernel(const int* __restrict__ src, const int* __restrict__ dst,
                                const float* __restrict__ xin, const float* __restrict__ tptr,
                                unsigned* __restrict__ m_u, long long EQ) {
    long long g = (long long)blockIdx.x * 256 + threadIdx.x;
    if (g >= EQ) return;
    int e = (int)(g >> 4), q = (int)(g & 15);
    int s = src[e], d = dst[e];
    float t = tptr[0];
    const float4 v = *(const float4*)(xin + (size_t)s * 64 + q * 4);
    unsigned* mp = m_u + (size_t)d * 64 + q * 4;
    atomicMax(mp + 0, fenc((fmaxf(v.x, 0.f) + EPS_GEN) * t));
    atomicMax(mp + 1, fenc((fmaxf(v.y, 0.f) + EPS_GEN) * t));
    atomicMax(mp + 2, fenc((fmaxf(v.z, 0.f) + EPS_GEN) * t));
    atomicMax(mp + 3, fenc((fmaxf(v.w, 0.f) + EPS_GEN) * t));
}

// ---------------------------------------------------------------------------
// Edge pass 2 (fused): denom += exp(z-m); agg += msg*exp(z-m)
// ---------------------------------------------------------------------------
__global__ void edge_sum_kernel(const int* __restrict__ src, const int* __restrict__ dst,
                                const float* __restrict__ xin, const float* __restrict__ tptr,
                                const unsigned* __restrict__ m_u,
                                float* __restrict__ denom, float* __restrict__ agg,
                                long long EQ) {
    long long g = (long long)blockIdx.x * 256 + threadIdx.x;
    if (g >= EQ) return;
    int e = (int)(g >> 4), q = (int)(g & 15);
    int s = src[e], d = dst[e];
    float t = tptr[0];
    const float4 v = *(const float4*)(xin + (size_t)s * 64 + q * 4);
    size_t base = (size_t)d * 64 + q * 4;
    const uint4 mu = *(const uint4*)(m_u + base);
    float m0 = fmaxf(v.x, 0.f) + EPS_GEN;
    float m1 = fmaxf(v.y, 0.f) + EPS_GEN;
    float m2 = fmaxf(v.z, 0.f) + EPS_GEN;
    float m3 = fmaxf(v.w, 0.f) + EPS_GEN;
    float e0 = __expf(m0 * t - fdec(mu.x));
    float e1 = __expf(m1 * t - fdec(mu.y));
    float e2 = __expf(m2 * t - fdec(mu.z));
    float e3 = __expf(m3 * t - fdec(mu.w));
    atomicAdd(denom + base + 0, e0);
    atomicAdd(denom + base + 1, e1);
    atomicAdd(denom + base + 2, e2);
    atomicAdd(denom + base + 3, e3);
    atomicAdd(agg + base + 0, m0 * e0);
    atomicAdd(agg + base + 1, m1 * e1);
    atomicAdd(agg + base + 2, m2 * e2);
    atomicAdd(agg + base + 3, m3 * e3);
}

// ---------------------------------------------------------------------------
// Node finalize: agg = agg/(denom+1e-16) + xin   (softmax normalize + root)
// ---------------------------------------------------------------------------
__global__ void finalize_kernel(float4* __restrict__ agg, const float4* __restrict__ denom,
                                const float4* __restrict__ xin, long long NH4) {
    long long g = (long long)blockIdx.x * 256 + threadIdx.x;
    if (g >= NH4) return;
    float4 a = agg[g], dn = denom[g], x = xin[g];
    a.x = a.x / (dn.x + 1e-16f) + x.x;
    a.y = a.y / (dn.y + 1e-16f) + x.y;
    a.z = a.z / (dn.z + 1e-16f) + x.z;
    a.w = a.w / (dn.w + 1e-16f) + x.w;
    agg[g] = a;
}

// ---------------------------------------------------------------------------
// GENConv MLP: out = relu(LN(in@W1 + b1; g1,be1)) @ W2 + b2 [+ residual]
// in [N,64], W1T f16 [128][64], W2T f16 [64][128]. block=128 thr, 16 nodes.
// ---------------------------------------------------------------------------
__global__ void mlp_kernel(const float* __restrict__ in,
                           const _Float16* __restrict__ W1T, const float* __restrict__ b1,
                           const float* __restrict__ g1, const float* __restrict__ be1,
                           const _Float16* __restrict__ W2T, const float* __restrict__ b2,
                           const float* __restrict__ residual,
                           float* __restrict__ out, int N) {
    __shared__ __align__(16) _Float16 aLds[16][64];
    __shared__ float    midF[16][128];
    __shared__ __align__(16) _Float16 midH[16][128];
    __shared__ float    red[16][16];
    __shared__ float    mu[16], rstd[16];

    int tid = threadIdx.x, wave = tid >> 5, lane = tid & 31;
    int node0 = blockIdx.x * 16;

    for (int i = tid; i < 16 * 64; i += 128) {
        int r = i >> 6, k = i & 63;
        int node = node0 + r;
        float v = (node < N) ? in[(size_t)node * 64 + k] : 0.f;
        aLds[r][k] = (_Float16)v;
    }
    __syncthreads();

    // GEMM1: wave covers cols [wave*32, wave*32+32); K=64 -> 2 steps
    v8f acc0 = {}, acc1 = {};
#pragma unroll
    for (int kk = 0; kk < 2; kk++) {
        v16h a  = load_frag(&aLds[0][0], 64, 0, kk * 32, lane);
        v16h b0 = load_frag(W1T, 64, wave * 32,      kk * 32, lane);
        v16h b1f= load_frag(W1T, 64, wave * 32 + 16, kk * 32, lane);
        acc0 = wmma16(a, b0, acc0);
        acc1 = wmma16(a, b1f, acc1);
    }
    {
        int hi = lane >> 4, cb = lane & 15;
        int c0 = wave * 32 + cb, c1 = wave * 32 + 16 + cb;
        float bb0 = b1[c0], bb1 = b1[c1];
#pragma unroll
        for (int r = 0; r < 8; r++) {
            int row = hi * 8 + r;
            midF[row][c0] = acc0[r] + bb0;
            midF[row][c1] = acc1[r] + bb1;
        }
    }
    __syncthreads();

    // LayerNorm(128) per row: 16 rows x 8 segments of 16
    {
        int row = tid >> 3, seg = tid & 7;
        float s = 0.f, q = 0.f;
#pragma unroll
        for (int j = 0; j < 16; j++) {
            float v = midF[row][seg * 16 + j];
            s += v; q += v * v;
        }
        red[row][seg] = s; red[row][8 + seg] = q;
    }
    __syncthreads();
    if (tid < 16) {
        float s = 0.f, q = 0.f;
#pragma unroll
        for (int j = 0; j < 8; j++) { s += red[tid][j]; q += red[tid][8 + j]; }
        float m = s * (1.f / 128.f);
        float v = q * (1.f / 128.f) - m * m;
        mu[tid] = m; rstd[tid] = rsqrtf(v + LN_EPS);
    }
    __syncthreads();
    {
        int row = tid >> 3, seg = tid & 7;
        float m = mu[row], rs = rstd[row];
#pragma unroll
        for (int j = 0; j < 16; j++) {
            int c = seg * 16 + j;
            float v = (midF[row][c] - m) * rs * g1[c] + be1[c];
            midH[row][c] = (_Float16)fmaxf(v, 0.f);
        }
    }
    __syncthreads();

    // GEMM2: wave -> 16-col tile; K=128 -> 4 steps
    v8f acc = {};
#pragma unroll
    for (int kk = 0; kk < 4; kk++) {
        v16h a = load_frag(&midH[0][0], 128, 0, kk * 32, lane);
        v16h b = load_frag(W2T, 128, wave * 16, kk * 32, lane);
        acc = wmma16(a, b, acc);
    }
    {
        int cb  = lane & 15;
        int col = wave * 16 + cb;
        int rbase = node0 + (lane >> 4) * 8;
        float bias = b2[col];
        if (node0 + 16 <= N) {
            if (residual) {
#pragma unroll
                for (int r = 0; r < 8; r++) {
                    size_t idx = (size_t)(rbase + r) * 64 + col;
                    out[idx] = acc[r] + bias + residual[idx];
                }
            } else {
#pragma unroll
                for (int r = 0; r < 8; r++)
                    out[(size_t)(rbase + r) * 64 + col] = acc[r] + bias;
            }
        } else {
#pragma unroll
            for (int r = 0; r < 8; r++) {
                if (rbase + r < N) {
                    size_t idx = (size_t)(rbase + r) * 64 + col;
                    float v = acc[r] + bias;
                    if (residual) v += residual[idx];
                    out[idx] = v;
                }
            }
        }
    }
}

// ---------------------------------------------------------------------------
// out = relu(LN(in; g,b)) over H=64 rows. 4 nodes / 256-thread block.
// group of 64 threads = 2 waves; shuffle butterfly + LDS cross-wave combine.
// ---------------------------------------------------------------------------
__global__ void ln_relu_kernel(const float* __restrict__ in, const float* __restrict__ g,
                               const float* __restrict__ b, float* __restrict__ out, int N) {
    __shared__ float ps[4][2][2];
    int tid = threadIdx.x;
    int grp = tid >> 6, wv = (tid >> 5) & 1, f = tid & 63;
    int node = blockIdx.x * 4 + grp;
    float v = (node < N) ? in[(size_t)node * 64 + f] : 0.f;
    float s = wave_sum(v);
    float q = wave_sum(v * v);
    if ((tid & 31) == 0) { ps[grp][wv][0] = s; ps[grp][wv][1] = q; }
    __syncthreads();
    float S = ps[grp][0][0] + ps[grp][1][0];
    float Q = ps[grp][0][1] + ps[grp][1][1];
    float m  = S * (1.f / 64.f);
    float va = Q * (1.f / 64.f) - m * m;
    float y  = (v - m) * rsqrtf(va + LN_EPS) * g[f] + b[f];
    if (node < N) out[(size_t)node * 64 + f] = fmaxf(y, 0.f);
}

// ---------------------------------------------------------------------------
// Final: t = relu(LN(in)); logit = t . W_lin + b_lin; out[i]=sigmoid, out[N+i]=logit
// ---------------------------------------------------------------------------
__global__ void final_kernel(const float* __restrict__ in, const float* __restrict__ g,
                             const float* __restrict__ b, const float* __restrict__ wlin,
                             const float* __restrict__ blin, float* __restrict__ out, int N) {
    __shared__ float ps[4][2][2];
    __shared__ float pp[4][2];
    int tid = threadIdx.x;
    int grp = tid >> 6, wv = (tid >> 5) & 1, f = tid & 63;
    int node = blockIdx.x * 4 + grp;
    float v = (node < N) ? in[(size_t)node * 64 + f] : 0.f;
    float s = wave_sum(v);
    float q = wave_sum(v * v);
    if ((tid & 31) == 0) { ps[grp][wv][0] = s; ps[grp][wv][1] = q; }
    __syncthreads();
    float S = ps[grp][0][0] + ps[grp][1][0];
    float Q = ps[grp][0][1] + ps[grp][1][1];
    float m  = S * (1.f / 64.f);
    float va = Q * (1.f / 64.f) - m * m;
    float y  = fmaxf((v - m) * rsqrtf(va + LN_EPS) * g[f] + b[f], 0.f);
    float p  = wave_sum(y * wlin[f]);
    if ((tid & 31) == 0) pp[grp][wv] = p;
    __syncthreads();
    if (f == 0 && node < N) {
        float logit = pp[grp][0] + pp[grp][1] + blin[0];
        out[node]     = 1.f / (1.f + __expf(-logit));
        out[N + node] = logit;
    }
}

// ---------------------------------------------------------------------------
// Host orchestration
// ---------------------------------------------------------------------------
extern "C" void kernel_launch(void* const* d_in, const int* in_sizes, int n_in,
                              void* d_out, int out_size, void* d_ws, size_t ws_size,
                              hipStream_t stream) {
    const float* x      = (const float*)d_in[0];
    const int*   eidx   = (const int*)d_in[1];
    const float* W_enc  = (const float*)d_in[2];
    const float* b_enc  = (const float*)d_in[3];
    const float* t1     = (const float*)d_in[4];
    const float* W1a    = (const float*)d_in[5];
    const float* b1a    = (const float*)d_in[6];
    const float* g1a    = (const float*)d_in[7];
    const float* be1a   = (const float*)d_in[8];
    const float* W1b    = (const float*)d_in[9];
    const float* b1b    = (const float*)d_in[10];
    const float* g_n1   = (const float*)d_in[11];
    const float* b_n1   = (const float*)d_in[12];
    const float* t2     = (const float*)d_in[13];
    const float* W2a    = (const float*)d_in[14];
    const float* b2a    = (const float*)d_in[15];
    const float* g2a    = (const float*)d_in[16];
    const float* be2a   = (const float*)d_in[17];
    const float* W2b    = (const float*)d_in[18];
    const float* b2b    = (const float*)d_in[19];
    const float* g_n0   = (const float*)d_in[20];
    const float* b_n0   = (const float*)d_in[21];
    const float* W_lin  = (const float*)d_in[22];
    const float* b_lin  = (const float*)d_in[23];
    float* outp = (float*)d_out;

    const int N = in_sizes[0] / 16;
    const int E = in_sizes[1] / 2;
    const int* src = eidx;
    const int* dst = eidx + E;

    const long long NH  = (long long)N * 64;
    const long long NH4 = NH / 4;
    const long long EQ  = (long long)E * 16;   // 4 features per thread

    // workspace layout (all L2-resident: 5 x 25.6MB + ~70KB f16 weights)
    float*    B0 = (float*)d_ws;          // h0 / xin2 / h3
    float*    B1 = B0 + NH;               // h2 (conv1 out), residual
    unsigned* B2 = (unsigned*)(B1 + NH);  // m (encoded)
    float*    B3 = (float*)B2 + NH;       // denom
    float*    B4 = B3 + NH;               // agg
    _Float16* WencT = (_Float16*)(B4 + NH);     // [64][32]
    _Float16* W1aT  = WencT + 64 * 32;          // [128][64]
    _Float16* W1bT  = W1aT + 128 * 64;          // [64][128]
    _Float16* W2aT  = W1bT + 64 * 128;          // [128][64]
    _Float16* W2bT  = W2aT + 128 * 64;          // [64][128]

    const int nodeBlocks = (N + 15) / 16;
    const int nh4Blocks  = (int)((NH4 + 255) / 256);
    const int eqBlocks   = (int)((EQ + 255) / 256);
    const int lnBlocks   = (N + 3) / 4;

    // weight prep (tiny)
    wconv_kernel<<<(64 * 32 + 255) / 256, 256, 0, stream>>>(W_enc, WencT, 16, 64, 32);
    wconv_kernel<<<(128 * 64 + 255) / 256, 256, 0, stream>>>(W1a, W1aT, 64, 128, 64);
    wconv_kernel<<<(64 * 128 + 255) / 256, 256, 0, stream>>>(W1b, W1bT, 128, 64, 128);
    wconv_kernel<<<(128 * 64 + 255) / 256, 256, 0, stream>>>(W2a, W2aT, 64, 128, 64);
    wconv_kernel<<<(64 * 128 + 255) / 256, 256, 0, stream>>>(W2b, W2bT, 128, 64, 128);

    // encoder: B0 = x @ W_enc + b_enc
    encoder_kernel<<<nodeBlocks, 128, 0, stream>>>(x, WencT, b_enc, B0, N);

    // ---- GENConv layer 1 (input B0) -> B1 ----
    init_kernel<<<nh4Blocks, 256, 0, stream>>>((uint4*)B2, (float4*)B3, (float4*)B4, NH4);
    edge_max_kernel<<<eqBlocks, 256, 0, stream>>>(src, dst, B0, t1, B2, EQ);
    edge_sum_kernel<<<eqBlocks, 256, 0, stream>>>(src, dst, B0, t1, B2, B3, B4, EQ);
    finalize_kernel<<<nh4Blocks, 256, 0, stream>>>((float4*)B4, (const float4*)B3,
                                                   (const float4*)B0, NH4);
    mlp_kernel<<<nodeBlocks, 128, 0, stream>>>(B4, W1aT, b1a, g1a, be1a,
                                               W1bT, b1b, nullptr, B1, N);

    // ---- layer 2: xin2 = relu(LN(h2)) -> B0; conv -> residual add -> B0 ----
    ln_relu_kernel<<<lnBlocks, 256, 0, stream>>>(B1, g_n1, b_n1, B0, N);
    init_kernel<<<nh4Blocks, 256, 0, stream>>>((uint4*)B2, (float4*)B3, (float4*)B4, NH4);
    edge_max_kernel<<<eqBlocks, 256, 0, stream>>>(src, dst, B0, t2, B2, EQ);
    edge_sum_kernel<<<eqBlocks, 256, 0, stream>>>(src, dst, B0, t2, B2, B3, B4, EQ);
    finalize_kernel<<<nh4Blocks, 256, 0, stream>>>((float4*)B4, (const float4*)B3,
                                                   (const float4*)B0, NH4);
    mlp_kernel<<<nodeBlocks, 128, 0, stream>>>(B4, W2aT, b2a, g2a, be2a,
                                               W2bT, b2b, /*residual=*/B1, B0, N);

    // ---- final LN + ReLU + linear + sigmoid ----
    final_kernel<<<lnBlocks, 256, 0, stream>>>(B0, g_n0, b_n0, W_lin, b_lin, outp, N);
}
